// Mamba_60516089200956
// MI455X (gfx1250) — compile-verified, matrix operand order checked by reading, not run on hardware
//
#include <hip/hip_runtime.h>
#include <hip/hip_bf16.h>

// ---------------- problem dimensions (fixed by the reference) ----------------
#define B_SZ    2
#define L_SZ    1024
#define DMODEL  2048
#define DSTATE  16
#define DCONV   4
#define DINNER  4096            // EXPAND * DMODEL
#define DTRANK  128             // ceil(DMODEL/16)
#define NT      (B_SZ * L_SZ)   // 2048 token rows
#define XPROJ_N (DTRANK + 2 * DSTATE)  // 160

// ---------------- WMMA / TDM vector types (gfx1250, wave32) ----------------
typedef __attribute__((ext_vector_type(16))) __bf16 v16bf;
typedef __attribute__((ext_vector_type(8)))  __bf16 v8bf;
typedef __attribute__((ext_vector_type(8)))  float  v8f;
typedef __attribute__((ext_vector_type(4)))  unsigned int u32x4;
typedef __attribute__((ext_vector_type(8)))  int i32x8;
typedef __attribute__((ext_vector_type(4)))  int i32x4;

#if __has_builtin(__builtin_amdgcn_tensor_load_to_lds) && \
    __has_builtin(__builtin_amdgcn_s_wait_tensorcnt)
#define HAVE_TDM 1
#else
#define HAVE_TDM 0
#endif

// round-to-nearest-even f32 -> bf16 (bit pattern), stored as ushort
__device__ __forceinline__ unsigned short f2bf(float f) {
    unsigned int u = __float_as_uint(f);
    unsigned int r = u + 0x7FFFu + ((u >> 16) & 1u);
    return (unsigned short)(r >> 16);
}

__device__ __forceinline__ float sigmoidf_(float v) {
    return 1.0f / (1.0f + __expf(-v));
}

#if HAVE_TDM
// Issue a TDM 2D tile load: tile_d1 rows x tile_d0 elements of 2-byte data,
// row stride = stride_elems, from global gaddr into LDS at lds_addr.
// D# layout per CDNA5 ISA ch.8 (group0/group1; groups 2/3 zero => dims 2+ unused).
__device__ __forceinline__ void tdm_load_2d(unsigned long long gaddr,
                                            unsigned lds_addr,
                                            unsigned tensor_d0, unsigned tensor_d1,
                                            unsigned tile_d0, unsigned tile_d1,
                                            unsigned long long stride_elems) {
    u32x4 g0;
    g0[0] = 1u;                                            // count=1 (valid D#)
    g0[1] = lds_addr;                                      // lds_addr [63:32]
    g0[2] = (unsigned)(gaddr & 0xFFFFFFFFull);             // global_addr lo
    g0[3] = (unsigned)((gaddr >> 32) & 0x01FFFFFFull)      // global_addr [56:32]
            | 0x80000000u;                                 // type=2 ("image")
    i32x8 g1;
    g1[0] = (int)(1u << 16);                               // data_size=1 (2B), wg_mask=0
    g1[1] = (int)((tensor_d0 & 0xFFFFu) << 16);            // tensor_dim0 lo16 @ [63:48]
    g1[2] = (int)((tensor_d0 >> 16) |                      // tensor_dim0 hi16
                  ((tensor_d1 & 0xFFFFu) << 16));          // tensor_dim1 lo16
    g1[3] = (int)((tensor_d1 >> 16) |                      // tensor_dim1 hi16
                  (tile_d0 << 16));                        // tile_dim0 @ [127:112]
    g1[4] = (int)(tile_d1 & 0xFFFFu);                      // tile_dim1 (tile_dim2=0)
    g1[5] = (int)(stride_elems & 0xFFFFFFFFull);           // dim0_stride lo32
    g1[6] = (int)((stride_elems >> 32) & 0xFFFFull);       // dim0_stride hi16
    g1[7] = 0;
    i32x4 gz = {0, 0, 0, 0};
#if defined(__clang_major__) && __clang_major__ >= 23
    i32x8 gz8 = {0, 0, 0, 0, 0, 0, 0, 0};
    __builtin_amdgcn_tensor_load_to_lds(g0, g1, gz, gz, gz8, 0);
#else
    __builtin_amdgcn_tensor_load_to_lds(g0, g1, gz, gz, 0);
#endif
}
#endif  // HAVE_TDM

// ---------------------------------------------------------------------------
// Fragment loader for 16-bit A/B operands, ISA 7.12.2 layout:
//   lanes 0-15  (lo): row M = lo, K halves {k..k+7} and {k+16..k+23}
//   lanes 16-31 (hi): row M = lo, K halves {k+8..k+15} and {k+24..k+31}
// Works for global or LDS source (addrspace inferred after inlining).
// ---------------------------------------------------------------------------
__device__ __forceinline__ v16bf load_frag16(const __bf16* base,
                                             int ld, int row, int k, int hi) {
    const __bf16* p = base + (size_t)row * ld + k + hi * 8;
    v8bf x0 = *(const v8bf*)(p);        // K = k + hi*8 + [0..7]
    v8bf x1 = *(const v8bf*)(p + 16);   // K = k + 16 + hi*8 + [0..7]
    v16bf r;
#pragma unroll
    for (int i = 0; i < 8; ++i) { r[i] = x0[i]; r[8 + i] = x1[i]; }
    return r;
}

// ---------------------------------------------------------------------------
// WMMA GEMM: C[M,N] = A[M,K] * B[N,K]^T   (B row-major as N x K)
// Each wave computes a 32x32 tile (4 accumulators, 4 v_wmma per K=32 step).
// Block = 256 threads = 8 waves stacked in M -> block tile 256 x 32.
// B slab (32 rows x 64 K-elements, 4KB bf16) is staged into LDS by the
// Tensor Data Mover, double-buffered, shared by all 8 waves (8x less global
// B traffic). A fragments stream from global (unique rows per wave, L2-hot).
// Requires M % 256 == 0, N % 32 == 0, K % 64 == 0 (true for all call sites).
// EPI: 0 = plain store; 1 = softplus(c + bias[n]) (dt_proj fusion).
// ---------------------------------------------------------------------------
template <int EPI>
__global__ __launch_bounds__(256) void gemm_wmma_bf16(
    const __bf16* __restrict__ A, const __bf16* __restrict__ Bw,
    float* __restrict__ C, const float* __restrict__ bias,
    int K, int N, int lda, int ldb, int ldc)
{
    const int lane = threadIdx.x & 31;
    const int wave = threadIdx.x >> 5;
    const int lo = lane & 15;
    const int hi = lane >> 4;

    const int m0 = blockIdx.y * 256 + wave * 32;  // this wave's 32 rows
    const int n0 = blockIdx.x * 32;               // this wave's 32 cols

    v8f c00 = {}, c01 = {}, c10 = {}, c11 = {};

#if HAVE_TDM
    __shared__ __align__(128) unsigned short bs[2][32][64];  // 2 x 4KB B slabs
    if (threadIdx.x == 0) {
        tdm_load_2d((unsigned long long)(uintptr_t)(Bw + (size_t)n0 * ldb),
                    (unsigned)(uintptr_t)&bs[0][0][0],
                    (unsigned)K, (unsigned)N, 64u, 32u,
                    (unsigned long long)ldb);
    }
    int buf = 0;
#endif

    for (int k = 0; k < K; k += 64) {
#if HAVE_TDM
        if (threadIdx.x == 0) __builtin_amdgcn_s_wait_tensorcnt(0);
        __syncthreads();  // slab `buf` resident & visible to all waves
        if (threadIdx.x == 0 && (k + 64 < K)) {
            // overlap: DMA the next slab while we compute on this one
            tdm_load_2d((unsigned long long)(uintptr_t)
                            (Bw + (size_t)n0 * ldb + (k + 64)),
                        (unsigned)(uintptr_t)&bs[buf ^ 1][0][0],
                        (unsigned)K, (unsigned)N, 64u, 32u,
                        (unsigned long long)ldb);
        }
        const __bf16* bsp = (const __bf16*)&bs[buf][0][0];
#endif
#pragma unroll
        for (int kk = 0; kk < 64; kk += 32) {
            if (k + kk + 32 < K) {  // uniform branch: EXEC all-ones at WMMA
                __builtin_prefetch(A + (size_t)(m0 + lo) * lda + k + kk + 32, 0, 1);
                __builtin_prefetch(A + (size_t)(m0 + 16 + lo) * lda + k + kk + 32, 0, 1);
            }
            v16bf a0 = load_frag16(A, lda, m0 + lo,      k + kk, hi);
            v16bf a1 = load_frag16(A, lda, m0 + 16 + lo, k + kk, hi);
#if HAVE_TDM
            v16bf b0 = load_frag16(bsp, 64, lo,      kk, hi);       // ds_load_b128
            v16bf b1 = load_frag16(bsp, 64, 16 + lo, kk, hi);
#else
            v16bf b0 = load_frag16(Bw, ldb, n0 + lo,      k + kk, hi);
            v16bf b1 = load_frag16(Bw, ldb, n0 + 16 + lo, k + kk, hi);
#endif
            c00 = __builtin_amdgcn_wmma_f32_16x16x32_bf16(false, a0, false, b0,
                                                          (short)0, c00, false, false);
            c01 = __builtin_amdgcn_wmma_f32_16x16x32_bf16(false, a0, false, b1,
                                                          (short)0, c01, false, false);
            c10 = __builtin_amdgcn_wmma_f32_16x16x32_bf16(false, a1, false, b0,
                                                          (short)0, c10, false, false);
            c11 = __builtin_amdgcn_wmma_f32_16x16x32_bf16(false, a1, false, b1,
                                                          (short)0, c11, false, false);
        }
#if HAVE_TDM
        __syncthreads();  // all waves done reading `buf` before it is re-filled
        buf ^= 1;
#endif
    }

    // C/D layout (ISA 7.12.2): VGPR r, lanes0-15 -> M=r, lanes16-31 -> M=8+r; N=lo
    auto store_tile = [&](const v8f& c, int mb, int nb) {
#pragma unroll
        for (int r = 0; r < 8; ++r) {
            const int row = mb + hi * 8 + r;
            const int col = nb + lo;
            float v = c[r];
            if constexpr (EPI == 1) {
                v += bias[col];
                v = (v > 20.0f) ? v : log1pf(__expf(v));  // softplus
            }
            C[(size_t)row * ldc + col] = v;
        }
    };
    store_tile(c00, m0,      n0);
    store_tile(c01, m0,      n0 + 16);
    store_tile(c10, m0 + 16, n0);
    store_tile(c11, m0 + 16, n0 + 16);
}

// ---------------- elementwise helpers ----------------
__global__ void cast_f32_to_bf16(const float* __restrict__ in,
                                 unsigned short* __restrict__ out, int n) {
    int i = blockIdx.x * blockDim.x + threadIdx.x;
    if (i < n) out[i] = f2bf(in[i]);
}

// depthwise causal conv (kernel 4) + bias + SiLU over x = xz[:, :DINNER]
__global__ void conv_silu_kernel(const float* __restrict__ xz,
                                 const float* __restrict__ cw,
                                 const float* __restrict__ cb,
                                 float* __restrict__ xconv,
                                 unsigned short* __restrict__ xconv_bf) {
    int idx = blockIdx.x * blockDim.x + threadIdx.x;   // NT * DINNER
    if (idx >= NT * DINNER) return;
    const int t = idx / DINNER, d = idx % DINNER;
    const int b = t / L_SZ, l = t % L_SZ;
    float acc = cb[d];
#pragma unroll
    for (int j = 0; j < DCONV; ++j) {
        const int ll = l - (DCONV - 1) + j;
        if (ll >= 0)
            acc += cw[d * DCONV + j] *
                   xz[((size_t)(b * L_SZ + ll)) * (2 * DINNER) + d];
    }
    const float s = acc * sigmoidf_(acc);
    xconv[idx] = s;
    xconv_bf[idx] = f2bf(s);
}

// pull dt_lo (first DTRANK cols of x_dbl, row stride XPROJ_N) out as bf16
__global__ void extract_dtlo(const float* __restrict__ xdbl,
                             unsigned short* __restrict__ dtlo) {
    int i = blockIdx.x * blockDim.x + threadIdx.x;   // NT * DTRANK
    if (i >= NT * DTRANK) return;
    const int t = i / DTRANK, r = i % DTRANK;
    dtlo[i] = f2bf(xdbl[(size_t)t * XPROJ_N + r]);
}

// selective scan: one thread per (b,d) channel, h[16] in registers,
// B_t / C_t broadcast through LDS once per timestep.
__global__ __launch_bounds__(256) void scan_kernel(
    const float* __restrict__ delta,   // [NT, DINNER]
    const float* __restrict__ xdbl,    // [NT, XPROJ_N] (B at 128, C at 144)
    const float* __restrict__ xconv,   // [NT, DINNER]
    const float* __restrict__ A_log,   // [DINNER, DSTATE]
    float* __restrict__ y)             // [NT, DINNER]
{
    __shared__ float Bs[DSTATE];
    __shared__ float Cs[DSTATE];
    const int d = blockIdx.x * 256 + threadIdx.x;
    const int b = blockIdx.y;

    float Arow[DSTATE];
#pragma unroll
    for (int n = 0; n < DSTATE; ++n)
        Arow[n] = -__expf(A_log[(size_t)d * DSTATE + n]);

    float h[DSTATE];
#pragma unroll
    for (int n = 0; n < DSTATE; ++n) h[n] = 0.0f;

    for (int l = 0; l < L_SZ; ++l) {
        const size_t t = (size_t)b * L_SZ + l;
        __syncthreads();  // protect Bs/Cs from previous iteration's readers
        if (threadIdx.x < 2 * DSTATE) {
            const float v = xdbl[t * XPROJ_N + DTRANK + threadIdx.x];
            if (threadIdx.x < DSTATE) Bs[threadIdx.x] = v;
            else                      Cs[threadIdx.x - DSTATE] = v;
        }
        __syncthreads();

        const float dt = delta[t * DINNER + d];
        const float xv = xconv[t * DINNER + d];
        const float dtx = dt * xv;
        float acc = 0.0f;
#pragma unroll
        for (int n = 0; n < DSTATE; ++n) {
            const float dA = __expf(dt * Arow[n]);   // v_exp_f32
            h[n] = h[n] * dA + dtx * Bs[n];
            acc = fmaf(h[n], Cs[n], acc);
        }
        y[t * DINNER + d] = acc;
    }
}

// y = (y_scan + xconv * D) * silu(z), emitted as bf16 for the out_proj GEMM
__global__ void gate_kernel(const float* __restrict__ y,
                            const float* __restrict__ xconv,
                            const float* __restrict__ xz,
                            const float* __restrict__ Dvec,
                            unsigned short* __restrict__ ybf) {
    int i = blockIdx.x * blockDim.x + threadIdx.x;   // NT * DINNER
    if (i >= NT * DINNER) return;
    const int t = i / DINNER, d = i % DINNER;
    const float z = xz[(size_t)t * (2 * DINNER) + DINNER + d];
    float v = y[i] + xconv[i] * Dvec[d];
    v *= z * sigmoidf_(z);
    ybf[i] = f2bf(v);
}

// ---------------------------------------------------------------------------
extern "C" void kernel_launch(void* const* d_in, const int* in_sizes, int n_in,
                              void* d_out, int out_size, void* d_ws, size_t ws_size,
                              hipStream_t stream) {
    (void)in_sizes; (void)n_in; (void)out_size; (void)ws_size;

    const float* hs       = (const float*)d_in[0];  // [B,L,DMODEL]
    const float* w_in     = (const float*)d_in[1];  // [2*DINNER, DMODEL]
    const float* conv_w   = (const float*)d_in[2];  // [DINNER, DCONV]
    const float* conv_b   = (const float*)d_in[3];  // [DINNER]
    const float* w_xproj  = (const float*)d_in[4];  // [XPROJ_N, DINNER]
    const float* w_dt     = (const float*)d_in[5];  // [DINNER, DTRANK]
    const float* b_dt     = (const float*)d_in[6];  // [DINNER]
    const float* A_log    = (const float*)d_in[7];  // [DINNER, DSTATE]
    const float* Dvec     = (const float*)d_in[8];  // [DINNER]
    const float* w_out    = (const float*)d_in[9];  // [DMODEL, DINNER]
    float* out = (float*)d_out;                     // [B,L,DMODEL] f32

    // ---- workspace carve-up (256B aligned) ----
    char* p = (char*)d_ws;
    auto carve = [&](size_t bytes) {
        char* r = p;
        p += (bytes + 255) & ~(size_t)255;
        return r;
    };
    unsigned short* hs_bf    = (unsigned short*)carve((size_t)NT * DMODEL * 2);
    unsigned short* win_bf   = (unsigned short*)carve((size_t)2 * DINNER * DMODEL * 2);
    unsigned short* wxp_bf   = (unsigned short*)carve((size_t)XPROJ_N * DINNER * 2);
    unsigned short* wdt_bf   = (unsigned short*)carve((size_t)DINNER * DTRANK * 2);
    unsigned short* wout_bf  = (unsigned short*)carve((size_t)DMODEL * DINNER * 2);
    float*          xz       = (float*)carve((size_t)NT * 2 * DINNER * 4);
    float*          xconv    = (float*)carve((size_t)NT * DINNER * 4);
    unsigned short* xconv_bf = (unsigned short*)carve((size_t)NT * DINNER * 2);
    float*          xdbl     = (float*)carve((size_t)NT * XPROJ_N * 4);
    unsigned short* dtlo_bf  = (unsigned short*)carve((size_t)NT * DTRANK * 2);
    float*          delta    = (float*)carve((size_t)NT * DINNER * 4);
    float*          yscan    = (float*)carve((size_t)NT * DINNER * 4);
    unsigned short* y_bf     = (unsigned short*)carve((size_t)NT * DINNER * 2);

    const int TB = 256;
    auto blocks = [&](size_t n) { return (unsigned)((n + TB - 1) / TB); };

    // 1) bf16 casts of GEMM operands
    cast_f32_to_bf16<<<blocks((size_t)NT * DMODEL), TB, 0, stream>>>(hs, hs_bf, NT * DMODEL);
    cast_f32_to_bf16<<<blocks((size_t)2 * DINNER * DMODEL), TB, 0, stream>>>(w_in, win_bf, 2 * DINNER * DMODEL);
    cast_f32_to_bf16<<<blocks((size_t)XPROJ_N * DINNER), TB, 0, stream>>>(w_xproj, wxp_bf, XPROJ_N * DINNER);
    cast_f32_to_bf16<<<blocks((size_t)DINNER * DTRANK), TB, 0, stream>>>(w_dt, wdt_bf, DINNER * DTRANK);
    cast_f32_to_bf16<<<blocks((size_t)DMODEL * DINNER), TB, 0, stream>>>(w_out, wout_bf, DMODEL * DINNER);

    // 2) in_proj GEMM: xz[NT, 8192] = hs[NT,2048] * w_in[8192,2048]^T
    gemm_wmma_bf16<0><<<dim3(2 * DINNER / 32, NT / 256), TB, 0, stream>>>(
        (const __bf16*)hs_bf, (const __bf16*)win_bf, xz, nullptr,
        DMODEL, 2 * DINNER, DMODEL, DMODEL, 2 * DINNER);

    // 3) depthwise causal conv + SiLU
    conv_silu_kernel<<<blocks((size_t)NT * DINNER), TB, 0, stream>>>(
        xz, conv_w, conv_b, xconv, xconv_bf);

    // 4) x_proj GEMM: xdbl[NT,160] = xconv[NT,4096] * w_xproj[160,4096]^T
    gemm_wmma_bf16<0><<<dim3(XPROJ_N / 32, NT / 256), TB, 0, stream>>>(
        (const __bf16*)xconv_bf, (const __bf16*)wxp_bf, xdbl, nullptr,
        DINNER, XPROJ_N, DINNER, DINNER, XPROJ_N);

    // 5) dt_lo -> bf16
    extract_dtlo<<<blocks((size_t)NT * DTRANK), TB, 0, stream>>>(xdbl, dtlo_bf);

    // 6) dt_proj GEMM fused with softplus(. + dt_proj_b)
    gemm_wmma_bf16<1><<<dim3(DINNER / 32, NT / 256), TB, 0, stream>>>(
        (const __bf16*)dtlo_bf, (const __bf16*)wdt_bf, delta, b_dt,
        DTRANK, DINNER, DTRANK, DTRANK, DINNER);

    // 7) selective scan (sequential in L, parallel over 8192 channels)
    scan_kernel<<<dim3(DINNER / 256, B_SZ), TB, 0, stream>>>(
        delta, xdbl, xconv, A_log, yscan);

    // 8) D-skip + SiLU(z) gating -> bf16
    gate_kernel<<<blocks((size_t)NT * DINNER), TB, 0, stream>>>(
        yscan, xconv, xz, Dvec, y_bf);

    // 9) out_proj GEMM: out[NT,2048] = y[NT,4096] * w_out[2048,4096]^T
    gemm_wmma_bf16<0><<<dim3(DMODEL / 32, NT / 256), TB, 0, stream>>>(
        (const __bf16*)y_bf, (const __bf16*)wout_bf, out, nullptr,
        DINNER, DMODEL, DINNER, DINNER, DMODEL);
}
